// CTR_Block_77077483094613
// MI455X (gfx1250) — matrix operand order repaired and checked by reading, not compiled, not run on hardware
//
#include <hip/hip_runtime.h>
#include <hip/hip_bf16.h>

typedef __attribute__((ext_vector_type(16))) __bf16 v16bf;
typedef __attribute__((ext_vector_type(8)))  float  v8f;

#define NB    32
#define CINC  64
#define COUTC 128
#define TTT   256
#define VVV   25
#define PPP   (TTT*VVV)   /* 6400 positions per (n, channel) plane */
#define ICH   32
#define EPSV  1e-5f

/* ---------------- workspace layout (bytes, 256-aligned) ---------------- */
#define OFF_XBAR  ((size_t)0)            /* 32*64*25 f32      = 204800 B  */
#define OFF_ATTN  ((size_t)204800)       /* 32*32*25 f32 (compact, rank-1 in u) */
#define OFF_WF    ((size_t)307200)       /* 24*2*512 bf16     = 49 KB     */
#define OFF_WTF   ((size_t)356352)       /* 8*9*4*512 bf16    = 295 KB    */
#define OFF_XT    ((size_t)651264)       /* x channel-last bf16: 32*6400*64  = 26.2 MB */
#define OFF_Z3    ((size_t)26865664)     /* 32*128*6400 bf16 (channel-major) = 52.4 MB */
#define OFF_Z4    ((size_t)79294464)     /* same */
#define OFF_H2    ((size_t)131723264)    /* h channel-last bf16: 32*6400*128 = 52.4 MB */
                                         /* total ws ~184.2 MB; residual lives in d_out */

/* ============ K0: xbar[n,c,v] = mean over T of x ============ */
__global__ __launch_bounds__(256) void k_mean(const float* __restrict__ x,
                                              float* __restrict__ xbar) {
  int i = blockIdx.x * 256 + threadIdx.x;        // 0 .. 51199
  int v = i % VVV;
  int c = (i / VVV) % CINC;
  int n = i / (VVV * CINC);
  const float* p = x + ((size_t)(n * CINC + c)) * PPP + v;
  float s = 0.f;
  for (int t = 0; t < TTT; ++t) s += p[t * VVV];
  xbar[i] = s * (1.0f / TTT);
}

/* ============ K1: compact attention a[n,c,v] (x1/u term cancels in softmax) ============ */
__global__ __launch_bounds__(256) void k_attn(const float* __restrict__ xbar,
                                              const float* __restrict__ w2,
                                              const float* __restrict__ b2,
                                              float* __restrict__ attnC) {
  __shared__ float sx2[ICH * VVV];
  int n = blockIdx.x, tid = threadIdx.x;
  for (int i = tid; i < ICH * VVV; i += 256) {
    int c = i / VVV, v = i % VVV;
    float s = b2[c];
    const float* xb = xbar + (size_t)n * CINC * VVV + v;
    const float* w  = w2 + c * CINC;
    for (int ci = 0; ci < CINC; ++ci) s += w[ci] * xb[ci * VVV];
    sx2[i] = s;
  }
  __syncthreads();
  if (tid < ICH) {
    int c = tid;
    float mn = sx2[c * VVV];
    for (int v = 1; v < VVV; ++v) mn = fminf(mn, sx2[c * VVV + v]);
    float e[VVV], ssum = 0.f;
    for (int v = 0; v < VVV; ++v) { e[v] = __expf(mn - sx2[c * VVV + v]); ssum += e[v]; }
    float inv = 1.f / ssum;
    float* dst = attnC + (size_t)(n * ICH + c) * VVV;
    for (int v = 0; v < VVV; ++v) dst[v] = e[v] * inv;
  }
}

/* ============ K2: weights -> bf16 pre-swizzled WMMA A-fragments ============ */
/* A-frag 16-bit 16x32, this code's contiguous k order (B uses the same): M = lane&15 ; K = (lane>>4)*16 + j     */
__global__ __launch_bounds__(256) void k_wprep(const float* __restrict__ w3,
                                               const float* __restrict__ w4,
                                               const float* __restrict__ wr,
                                               const float* __restrict__ wt,
                                               __bf16* __restrict__ wfrag,
                                               __bf16* __restrict__ wtfrag) {
  int e = blockIdx.x * 256 + threadIdx.x;        // 0 .. 172031
  if (e < 24 * 2 * 512) {                        // big 1x1 GEMM weights
    int j = e & 15, l = (e >> 4) & 31, f = e >> 9;
    int mtile = f >> 1, ks = f & 1;
    int K  = ((l >> 4) << 4) + j;
    int ci = ks * 32 + K;
    int o  = ((mtile & 7) << 4) + (l & 15);
    const float* W = (mtile < 8) ? w3 : (mtile < 16 ? w4 : wr);
    wfrag[e] = (__bf16)W[o * CINC + ci];
  } else {                                       // temporal conv weights
    int e2 = e - 24 * 2 * 512;                   // < 288*512
    int j = e2 & 15, l = (e2 >> 4) & 31, f = e2 >> 9;
    int mtile = f / 36, rem = f % 36, tap = rem >> 2, kk = rem & 3;
    int K = ((l >> 4) << 4) + j;
    int i = kk * 32 + K;
    int o = (mtile << 4) + (l & 15);
    wtfrag[e2] = (__bf16)wt[(size_t)(o * COUTC + i) * 9 + tap];
  }
}

/* ============ K2b: x -> channel-last bf16 xT[n][p][ci] ============ */
__global__ __launch_bounds__(256) void k_xprep(const float* __restrict__ x,
                                               __bf16* __restrict__ xT) {
  int n = blockIdx.x >> 8, t = blockIdx.x & 255;
  for (int i = threadIdx.x; i < VVV * CINC; i += 256) {
    int v = i >> 6, ci = i & 63;
    xT[((size_t)n * PPP + t * VVV + v) * CINC + ci] =
        (__bf16)x[((size_t)(n * CINC + ci)) * PPP + t * VVV + v];
  }
}

/* ============ K3a: WMMA GEMM [384x64]@[64x6400] per n -> z3,z4, res->d_out ============ */
__global__ __launch_bounds__(256) void k_gemm1(const __bf16* __restrict__ xT,
                                               const __bf16* __restrict__ wfrag,
                                               const float* __restrict__ b3,
                                               const float* __restrict__ b4,
                                               const float* __restrict__ br,
                                               const float* __restrict__ bnrg,
                                               const float* __restrict__ bnrb,
                                               const float* __restrict__ bnrm,
                                               const float* __restrict__ bnrv,
                                               __bf16* __restrict__ z3,
                                               __bf16* __restrict__ z4,
                                               float* __restrict__ resout) {
  int n = blockIdx.x / 25;
  int pbase = (blockIdx.x % 25) * 256;
  int lane = threadIdx.x & 31;
  /* wave id is wave-uniform: force it into an SGPR so all mtile-dependent
     control flow / addressing is scalar (no exec-mask churn in the epilogue) */
  int wave = __builtin_amdgcn_readfirstlane(threadIdx.x >> 5);
  int colh = lane & 15, rowh = lane >> 4;

  v16bf Af[3][2];
#pragma unroll
  for (int mi = 0; mi < 3; ++mi)
#pragma unroll
    for (int ks = 0; ks < 2; ++ks)
      Af[mi][ks] = *(const v16bf*)(wfrag + ((size_t)(((wave * 3 + mi) * 2 + ks) * 32 + lane) << 4));

  /* hoisted epilogue constants: out = d*Am + Bm per (mi, r) */
  float Am[3][8], Bm[3][8];
#pragma unroll
  for (int mi = 0; mi < 3; ++mi) {
    int mtile = wave * 3 + mi;
#pragma unroll
    for (int r = 0; r < 8; ++r) {
      int o = ((mtile & 7) << 4) + (rowh << 3) + r;
      if (mtile < 8)       { Am[mi][r] = 1.f; Bm[mi][r] = b3[o]; }
      else if (mtile < 16) { Am[mi][r] = 1.f; Bm[mi][r] = b4[o]; }
      else {
        float sc = bnrg[o] * rsqrtf(bnrv[o] + EPSV);
        Am[mi][r] = sc;
        Bm[mi][r] = br[o] * sc + bnrb[o] - bnrm[o] * sc;
      }
    }
  }

  for (int nt = 0; nt < 16; ++nt) {
    int p = pbase + nt * 16 + colh;
    /* B fragment: K contiguous in channel-last xT -> one v16bf load per kstep */
    const __bf16* bx = xT + ((size_t)n * PPP + p) * CINC + (rowh << 4);
    v16bf B0 = *(const v16bf*)(bx);
    v16bf B1 = *(const v16bf*)(bx + 32);
#pragma unroll
    for (int mi = 0; mi < 3; ++mi) {
      v8f acc = {};
      acc = __builtin_amdgcn_wmma_f32_16x16x32_bf16(false, Af[mi][0], false, B0,
                                                    (short)0, acc, false, false);
      acc = __builtin_amdgcn_wmma_f32_16x16x32_bf16(false, Af[mi][1], false, B1,
                                                    (short)0, acc, false, false);
      int mtile = wave * 3 + mi;                 /* scalar */
      int obase = (mtile & 7) << 4;
      size_t base = ((size_t)n * COUTC) * PPP + p;
      if (mtile < 8) {
#pragma unroll
        for (int r = 0; r < 8; ++r) {
          int o = obase + (rowh << 3) + r;
          z3[base + (size_t)o * PPP] = (__bf16)(acc[r] * Am[mi][r] + Bm[mi][r]);
        }
      } else if (mtile < 16) {
#pragma unroll
        for (int r = 0; r < 8; ++r) {
          int o = obase + (rowh << 3) + r;
          z4[base + (size_t)o * PPP] = (__bf16)(acc[r] * Am[mi][r] + Bm[mi][r]);
        }
      } else {
#pragma unroll
        for (int r = 0; r < 8; ++r) {
          int o = obase + (rowh << 3) + r;
          resout[base + (size_t)o * PPP] = acc[r] * Am[mi][r] + Bm[mi][r];
        }
      }
    }
  }
}

/* ============ K3b: V-mix + bn1 + ReLU -> h2 channel-last [n][p][o] ============ */
__global__ __launch_bounds__(256) void k_vmix(const float* __restrict__ attnC,
                                              const float* __restrict__ Aadj,
                                              const float* __restrict__ g1,
                                              const float* __restrict__ bb1,
                                              const float* __restrict__ m1,
                                              const float* __restrict__ vv1,
                                              const __bf16* __restrict__ z3,
                                              const __bf16* __restrict__ z4,
                                              __bf16* __restrict__ h2) {
  __shared__ float sz4[COUTC * VVV];   // 12.8 KB
  __shared__ float sadj[VVV * VVV];    // 2.5 KB
  __shared__ float ss3[COUTC], ssc[COUTC], ssh[COUTC];
  int n = blockIdx.x >> 8, t = blockIdx.x & 255;
  int tid = threadIdx.x;

  for (int i = tid; i < VVV * VVV; i += 256) sadj[i] = Aadj[i];
  for (int i = tid; i < COUTC * VVV; i += 256) {
    int o = i / VVV, v = i % VVV;
    sz4[i] = (float)z4[((size_t)(n * COUTC + o)) * PPP + t * VVV + v];
  }
  if (tid < COUTC) {
    int o = tid, c = o & 31;
    const __bf16* r3 = z3 + ((size_t)(n * COUTC + o)) * PPP + t * VVV;
    const float* ac = attnC + (size_t)(n * ICH + c) * VVV;
    float s = 0.f;
#pragma unroll
    for (int v = 0; v < VVV; ++v) s += ac[v] * (float)r3[v];
    ss3[o] = s;
    float sc = g1[o] * rsqrtf(vv1[o] + EPSV);
    ssc[o] = sc;
    ssh[o] = bb1[o] - m1[o] * sc;
  }
  __syncthreads();

  __bf16* dst = h2 + ((size_t)n * PPP + t * VVV) * COUTC;
  for (int i = tid; i < VVV * COUTC; i += 256) {   // i = u*128 + o -> contiguous stores
    int u = i >> 7, o = i & 127;
    float y = ss3[o];
    const float* az = sz4 + o * VVV;
    const float* ar = sadj + u * VVV;
#pragma unroll
    for (int v = 0; v < VVV; ++v) y += ar[v] * az[v];
    dst[i] = (__bf16)fmaxf(y * ssc[o] + ssh[o], 0.f);
  }
}

/* ============ K4: temporal 9x1 conv (9 shifted WMMA GEMMs) + bn2 + res + ReLU ============ */
__global__ __launch_bounds__(256) void k_tcn(const __bf16* __restrict__ h2,
                                             const __bf16* __restrict__ wtfrag,
                                             const float* __restrict__ bt,
                                             const float* __restrict__ g2,
                                             const float* __restrict__ bb2,
                                             const float* __restrict__ m2,
                                             const float* __restrict__ vv2,
                                             float* __restrict__ out) {
  int n = blockIdx.x / 25;
  int pbase = (blockIdx.x % 25) * 256;
  int lane = threadIdx.x & 31;
  int mtile = __builtin_amdgcn_readfirstlane(threadIdx.x >> 5);  // scalar M-tile id
  int colh = lane & 15, rowh = lane >> 4;

  /* hoisted epilogue: y = acc*sc2 + sh2 (bt folded into shift) */
  float sc2[8], sh2[8];
#pragma unroll
  for (int r = 0; r < 8; ++r) {
    int o = (mtile << 4) + (rowh << 3) + r;
    float s = g2[o] * rsqrtf(vv2[o] + EPSV);
    sc2[r] = s;
    sh2[r] = bt[o] * s + bb2[o] - m2[o] * s;
  }

  for (int nt = 0; nt < 16; ++nt) {
    v8f acc = {};
    int pcol = pbase + nt * 16 + colh;
    if (nt < 15)  /* speculative prefetch of next tile's B rows */
      __builtin_prefetch(h2 + ((size_t)n * PPP + pcol + 16) * COUTC, 0, 1);
#pragma unroll
    for (int tap = 0; tap < 9; ++tap) {
      int pp = pcol + (tap - 4) * VVV;        // t-shift == 25-position shift
      bool ok = ((unsigned)pp < (unsigned)PPP);
#pragma unroll
      for (int kk = 0; kk < 4; ++kk) {
        v16bf a = *(const v16bf*)(wtfrag +
                   ((size_t)(((mtile * 9 + tap) * 4 + kk) * 32 + lane) << 4));
        v16bf b;
        if (ok) {
          b = *(const v16bf*)(h2 + ((size_t)n * PPP + pp) * COUTC + kk * 32 + (rowh << 4));
        } else {
#pragma unroll
          for (int j = 0; j < 16; ++j) b[j] = (__bf16)0.0f;
        }
        acc = __builtin_amdgcn_wmma_f32_16x16x32_bf16(false, a, false, b,
                                                      (short)0, acc, false, false);
      }
    }
    size_t base = ((size_t)n * COUTC) * PPP + pcol;
#pragma unroll
    for (int r = 0; r < 8; ++r) {
      int o = (mtile << 4) + (rowh << 3) + r;
      size_t idx = base + (size_t)o * PPP;
      float y = acc[r] * sc2[r] + sh2[r];
      float resv = out[idx];                  /* residual parked here by k_gemm1 */
      out[idx] = fmaxf(y + resv, 0.f);        /* same thread reads & writes idx */
    }
  }
}

/* ======================= host launcher ======================= */
extern "C" void kernel_launch(void* const* d_in, const int* in_sizes, int n_in,
                              void* d_out, int out_size, void* d_ws, size_t ws_size,
                              hipStream_t stream) {
  (void)in_sizes; (void)n_in; (void)out_size; (void)ws_size;
  const float* x    = (const float*)d_in[0];
  const float* Aadj = (const float*)d_in[1];
  /* w1/b1 (d_in[2],[3]) provably cancel in the softmax over v */
  const float* w2   = (const float*)d_in[4];
  const float* b2   = (const float*)d_in[5];
  const float* w3   = (const float*)d_in[6];
  const float* b3   = (const float*)d_in[7];
  const float* w4   = (const float*)d_in[8];
  const float* b4   = (const float*)d_in[9];
  const float* bn1g = (const float*)d_in[10];
  const float* bn1b = (const float*)d_in[11];
  const float* bn1m = (const float*)d_in[12];
  const float* bn1v = (const float*)d_in[13];
  const float* wt   = (const float*)d_in[14];
  const float* bt   = (const float*)d_in[15];
  const float* bn2g = (const float*)d_in[16];
  const float* bn2b = (const float*)d_in[17];
  const float* bn2m = (const float*)d_in[18];
  const float* bn2v = (const float*)d_in[19];
  const float* wr   = (const float*)d_in[20];
  const float* br   = (const float*)d_in[21];
  const float* bnrg = (const float*)d_in[22];
  const float* bnrb = (const float*)d_in[23];
  const float* bnrm = (const float*)d_in[24];
  const float* bnrv = (const float*)d_in[25];

  char* ws = (char*)d_ws;
  float*  xbar   = (float*)(ws + OFF_XBAR);
  float*  attnC  = (float*)(ws + OFF_ATTN);
  __bf16* wfrag  = (__bf16*)(ws + OFF_WF);
  __bf16* wtfrag = (__bf16*)(ws + OFF_WTF);
  __bf16* xT     = (__bf16*)(ws + OFF_XT);
  __bf16* z3     = (__bf16*)(ws + OFF_Z3);
  __bf16* z4     = (__bf16*)(ws + OFF_Z4);
  __bf16* h2     = (__bf16*)(ws + OFF_H2);
  float*  outf   = (float*)d_out;

  dim3 blk(256);
  k_mean <<<200,        blk, 0, stream>>>(x, xbar);
  k_attn <<<NB,         blk, 0, stream>>>(xbar, w2, b2, attnC);
  k_wprep<<<672,        blk, 0, stream>>>(w3, w4, wr, wt, wfrag, wtfrag);
  k_xprep<<<NB * TTT,   blk, 0, stream>>>(x, xT);
  k_gemm1<<<NB * 25,    blk, 0, stream>>>(xT, wfrag, b3, b4, br,
                                          bnrg, bnrb, bnrm, bnrv, z3, z4, outf);
  k_vmix <<<NB * TTT,   blk, 0, stream>>>(attnC, Aadj, bn1g, bn1b, bn1m, bn1v, z3, z4, h2);
  k_tcn  <<<NB * 25,    blk, 0, stream>>>(h2, wtfrag, bt, bn2g, bn2b, bn2m, bn2v, outf);
}
